// SingleStreamBlock_50388556316739
// MI455X (gfx1250) — matrix operand order repaired and checked by reading, not compiled, not run on hardware
//
#include <hip/hip_runtime.h>
#include <hip/hip_bf16.h>

#define L_SEQ    2048
#define DIM      3072
#define HEAD_DIM 128
#define N_HEADS  24
#define MLP_DIM  12288
#define W1_ROWS  (3*DIM + 2*MLP_DIM)   // 33792
#define W2_COLS  (DIM + MLP_DIM)       // 15360
#define QKV_W    (3*DIM)               // 9216

typedef __bf16 bf16;
typedef __attribute__((ext_vector_type(16))) __bf16 v16bf;
typedef __attribute__((ext_vector_type(8)))  __bf16 v8bf;
typedef __attribute__((ext_vector_type(8)))  float  v8f;

#if defined(__HIP_DEVICE_COMPILE__) && __has_builtin(__builtin_amdgcn_tensor_load_to_lds)
#define USE_TDM 1
#else
#define USE_TDM 0
#endif

// ---------------- WMMA helpers (CDNA5 wave32, 16x16x32 bf16 -> f32) ----------

__device__ __forceinline__ v8f wmma_bf16(v16bf a, v16bf b, v8f c) {
  return __builtin_amdgcn_wmma_f32_16x16x32_bf16(false, a, false, b, (short)0, c,
                                                 false, false);
}

__device__ __forceinline__ v8f zero_v8f() {
  v8f z;
#pragma unroll
  for (int i = 0; i < 8; ++i) z[i] = 0.f;
  return z;
}

// A-matrix 16x32 bf16 fragment from row-major [M x K] source (stride ld).
// ISA layout: lanes 0-15 hold M=lane, K=0..7 (v0-3) and K=16..23 (v4-7);
// lanes 16-31 hold K=8..15 and K=24..31.  -> two 16B contiguous loads per lane.
__device__ __forceinline__ v16bf load_a16x32(const bf16* base, int ld, int m0, int k0) {
  const int lane = threadIdx.x & 31;
  const int half = lane >> 4;
  const bf16* p = base + (size_t)(m0 + (lane & 15)) * ld + k0 + half * 8;
  v8bf lo = *(const v8bf*)p;
  v8bf hi = *(const v8bf*)(p + 16);
  v16bf r;
#pragma unroll
  for (int i = 0; i < 8; ++i) { r[i] = lo[i]; r[i + 8] = hi[i]; }
  return r;
}

// B-matrix 32x16 bf16 fragment.  B[k][n] is sourced from a transposed buffer
// where row n holds K contiguously (i.e. baseT is [N x K] row-major).
// ISA layout: lanes 0-15 hold N=lane, K=0..7 and K=16..23; lanes 16-31 hold K=8..15 and K=24..31.
__device__ __forceinline__ v16bf load_b32x16(const bf16* baseT, int ld, int n0, int k0) {
  const int lane = threadIdx.x & 31;
  const int half = lane >> 4;
  const bf16* p = baseT + (size_t)(n0 + (lane & 15)) * ld + k0 + half * 8;
  v8bf lo = *(const v8bf*)p;
  v8bf hi = *(const v8bf*)(p + 16);
  v16bf r;
#pragma unroll
  for (int i = 0; i < 8; ++i) { r[i] = lo[i]; r[i + 8] = hi[i]; }
  return r;
}

// ---------------- Tensor Data Mover: 64 rows x 64 bf16 tile -> LDS ----------
// D# per CDNA5 ISA: group0 = {count, lds_addr,
// global_addr, type=2}; group1 = {data_size=2B, tensor dims (huge, tiles are
// always in-bounds by construction), tile_dim0=64, tile_dim1=64, dim0 stride}.
// Higher-dim groups zero (2-D tensor, tile_dim2=0 -> unused).

__device__ __forceinline__ unsigned lds_off(const void* p) {
  return (unsigned)(unsigned long long)(uintptr_t)p;   // LDS aperture: offset in low 32 bits
}

__device__ __forceinline__ void tdm_load_tile64x64(unsigned ldsOff, const bf16* gsrc,
                                                   unsigned ldElems) {
#if USE_TDM
  typedef unsigned int v4u __attribute__((ext_vector_type(4)));
  typedef int v8i_ __attribute__((ext_vector_type(8)));
  typedef int v4i_ __attribute__((ext_vector_type(4)));
  const unsigned long long ga = (unsigned long long)(uintptr_t)gsrc;
  v4u g0;
  g0[0] = 1u;                                           // count=1 (valid user D#)
  g0[1] = ldsOff;                                       // lds_addr (bytes)
  g0[2] = (unsigned)ga;                                 // global_addr[31:0]
  g0[3] = (unsigned)((ga >> 32) & 0x01ffffffu) | (2u << 30);  // addr[56:32] | type=2
  v8i_ g1;
  g1[0] = (int)(1u << 16);                 // workgroup_mask=0, data_size=1 (2 bytes)
  g1[1] = (int)0xffff0000u;                // tensor_dim0[15:0]=0xFFFF (huge)
  g1[2] = (int)(0x7fffu | 0xffff0000u);    // tensor_dim0[31:16] | tensor_dim1[15:0]
  g1[3] = (int)(0x7fffu | (64u << 16));    // tensor_dim1[31:16] | tile_dim0=64
  g1[4] = 64;                              // tile_dim1=64, tile_dim2=0
  g1[5] = (int)ldElems;                    // tensor_dim0_stride[31:0] (elements)
  g1[6] = 0;                               // stride[47:32]=0, dim1_stride lo=0
  g1[7] = 0;
  v4i_ gz4;
  gz4[0] = 0; gz4[1] = 0; gz4[2] = 0; gz4[3] = 0;
  v8i_ gz8;
#pragma unroll
  for (int i = 0; i < 8; ++i) gz8[i] = 0;
  // 6-arg form (clang-23 / therock-10.0 headers): groups 0..3 + extra group + cpol
  __builtin_amdgcn_tensor_load_to_lds(g0, g1, gz4, gz4, gz8, 0);
#else
  (void)ldsOff; (void)gsrc; (void)ldElems;
#endif
}

__device__ __forceinline__ void tdm_wait0() {
#if USE_TDM
  __builtin_amdgcn_s_wait_tensorcnt(0);
#endif
}

// ---------------- 1) weight fp32 -> bf16 conversion --------------------------

__global__ __launch_bounds__(256) void wcvt_kernel(const float* __restrict__ w1,
                                                   const float* __restrict__ w2,
                                                   unsigned short* __restrict__ w1b_,
                                                   unsigned short* __restrict__ w2b_) {
  bf16* w1b = (bf16*)w1b_;
  bf16* w2b = (bf16*)w2b_;
  const size_t n1 = (size_t)W1_ROWS * DIM;
  const size_t n2 = (size_t)DIM * W2_COLS;
  const size_t stride = (size_t)gridDim.x * 256;
  for (size_t i = (size_t)blockIdx.x * 256 + threadIdx.x; i < n1; i += stride)
    w1b[i] = (bf16)w1[i];
  for (size_t i = (size_t)blockIdx.x * 256 + threadIdx.x; i < n2; i += stride)
    w2b[i] = (bf16)w2[i];
}

// ---------------- 2) LayerNorm + modulation -> bf16 activations --------------

__global__ __launch_bounds__(256) void ln_mod_kernel(const float* __restrict__ x,
                                                     const float* __restrict__ shift,
                                                     const float* __restrict__ scale,
                                                     unsigned short* __restrict__ xmod_) {
  bf16* xmod = (bf16*)xmod_;
  __shared__ float s_sum[256], s_sq[256];
  const int row = blockIdx.x;
  const float* xr = x + (size_t)row * DIM;
  float sum = 0.f, sq = 0.f;
  for (int c = threadIdx.x; c < DIM; c += 256) {
    float v = xr[c];
    sum += v;
    sq += v * v;
  }
  s_sum[threadIdx.x] = sum;
  s_sq[threadIdx.x] = sq;
  __syncthreads();
  for (int s = 128; s > 0; s >>= 1) {
    if (threadIdx.x < s) {
      s_sum[threadIdx.x] += s_sum[threadIdx.x + s];
      s_sq[threadIdx.x]  += s_sq[threadIdx.x + s];
    }
    __syncthreads();
  }
  const float mu  = s_sum[0] * (1.f / DIM);
  const float var = s_sq[0] * (1.f / DIM) - mu * mu;
  const float rs  = rsqrtf(var + 1e-6f);
  for (int c = threadIdx.x; c < DIM; c += 256) {
    float v = (xr[c] - mu) * rs * (1.f + scale[c]) + shift[c];
    xmod[(size_t)row * DIM + c] = (bf16)v;
  }
}

// ---------------- 3+8) TDM double-buffered GEMM core -------------------------
// Block = 128 threads (4 waves), 64x64 output tile; wave w owns the 32x32
// sub-tile at ((w>>1)*32, (w&1)*32).  Wave 0 DMAs the next A/B 64x64 tiles via
// TDM while all waves run 8 WMMAs each on the current LDS buffers (K-step 64).

template <int KTOT, bool RESIDUAL>
__device__ __forceinline__ void gemm_core(const bf16* __restrict__ A,
                                          const bf16* __restrict__ B,
                                          float* __restrict__ C, int ldc,
                                          const float* __restrict__ x,
                                          const float* __restrict__ gate) {
  __shared__ bf16 ldsA[2][64][64];   // 16 KB
  __shared__ bf16 ldsB[2][64][64];   // 16 KB
  const int wid = threadIdx.x >> 5;
  const int m0 = blockIdx.y * 64;
  const int n0 = blockIdx.x * 64;
  const int mSub = (wid >> 1) * 32;
  const int nSub = (wid & 1) * 32;
  v8f acc00 = zero_v8f(), acc01 = zero_v8f(), acc10 = zero_v8f(), acc11 = zero_v8f();

#if USE_TDM
  if (wid == 0) {
    tdm_load_tile64x64(lds_off(&ldsA[0][0][0]), A + (size_t)m0 * KTOT, KTOT);
    tdm_load_tile64x64(lds_off(&ldsB[0][0][0]), B + (size_t)n0 * KTOT, KTOT);
  }
  for (int k = 0, it = 0; k < KTOT; k += 64, ++it) {
    const int cur = it & 1;
    if (wid == 0) tdm_wait0();
    __syncthreads();
    if (wid == 0 && (k + 64) < KTOT) {
      tdm_load_tile64x64(lds_off(&ldsA[cur ^ 1][0][0]), A + (size_t)m0 * KTOT + k + 64, KTOT);
      tdm_load_tile64x64(lds_off(&ldsB[cur ^ 1][0][0]), B + (size_t)n0 * KTOT + k + 64, KTOT);
    }
#pragma unroll
    for (int kc = 0; kc < 2; ++kc) {
      v16bf a0 = load_a16x32(&ldsA[cur][0][0], 64, mSub, kc * 32);
      v16bf a1 = load_a16x32(&ldsA[cur][0][0], 64, mSub + 16, kc * 32);
      v16bf b0 = load_b32x16(&ldsB[cur][0][0], 64, nSub, kc * 32);
      v16bf b1 = load_b32x16(&ldsB[cur][0][0], 64, nSub + 16, kc * 32);
      acc00 = wmma_bf16(a0, b0, acc00);
      acc01 = wmma_bf16(a0, b1, acc01);
      acc10 = wmma_bf16(a1, b0, acc10);
      acc11 = wmma_bf16(a1, b1, acc11);
    }
    __syncthreads();
  }
#else
  for (int k = 0; k < KTOT; k += 32) {
    v16bf a0 = load_a16x32(A, KTOT, m0 + mSub, k);
    v16bf a1 = load_a16x32(A, KTOT, m0 + mSub + 16, k);
    v16bf b0 = load_b32x16(B, KTOT, n0 + nSub, k);
    v16bf b1 = load_b32x16(B, KTOT, n0 + nSub + 16, k);
    acc00 = wmma_bf16(a0, b0, acc00);
    acc01 = wmma_bf16(a0, b1, acc01);
    acc10 = wmma_bf16(a1, b0, acc10);
    acc11 = wmma_bf16(a1, b1, acc11);
  }
#endif

  const int lane = threadIdx.x & 31;
  const int half = lane >> 4, cn = lane & 15;
  const int mBase = m0 + mSub, nBase = n0 + nSub;
#pragma unroll
  for (int r = 0; r < 8; ++r) {
    const int r0 = mBase + r + half * 8;
    const int r1 = mBase + 16 + r + half * 8;
    const int c0 = nBase + cn;
    const int c1 = nBase + 16 + cn;
    if (RESIDUAL) {
      C[(size_t)r0 * ldc + c0] = x[(size_t)r0 * ldc + c0] + gate[c0] * acc00[r];
      C[(size_t)r0 * ldc + c1] = x[(size_t)r0 * ldc + c1] + gate[c1] * acc01[r];
      C[(size_t)r1 * ldc + c0] = x[(size_t)r1 * ldc + c0] + gate[c0] * acc10[r];
      C[(size_t)r1 * ldc + c1] = x[(size_t)r1 * ldc + c1] + gate[c1] * acc11[r];
    } else {
      C[(size_t)r0 * ldc + c0] = acc00[r];
      C[(size_t)r0 * ldc + c1] = acc01[r];
      C[(size_t)r1 * ldc + c0] = acc10[r];
      C[(size_t)r1 * ldc + c1] = acc11[r];
    }
  }
}

__global__ __launch_bounds__(128) void gemm1_kernel(const unsigned short* __restrict__ A_,
                                                    const unsigned short* __restrict__ B_,
                                                    float* __restrict__ C) {
  gemm_core<DIM, false>((const bf16*)A_, (const bf16*)B_, C, W1_ROWS, nullptr, nullptr);
}

__global__ __launch_bounds__(128) void gemm2_kernel(const unsigned short* __restrict__ A_,
                                                    const unsigned short* __restrict__ B_,
                                                    const float* __restrict__ x,
                                                    const float* __restrict__ gate,
                                                    float* __restrict__ Y) {
  gemm_core<W2_COLS, true>((const bf16*)A_, (const bf16*)B_, Y, DIM, x, gate);
}

// ---------------- 4) q/k RMSNorm + RoPE -> bf16 [head][L][128] ---------------

__global__ __launch_bounds__(128) void qkprep_kernel(const float* __restrict__ h,
                                                     const float* __restrict__ pe,
                                                     const float* __restrict__ q_scale,
                                                     const float* __restrict__ k_scale,
                                                     unsigned short* __restrict__ qb_,
                                                     unsigned short* __restrict__ kb_) {
  bf16* qb = (bf16*)qb_;
  bf16* kb = (bf16*)kb_;
  const int wid = threadIdx.x >> 5, lane = threadIdx.x & 31;
  const int task = blockIdx.x * 4 + wid;     // 2048 * 24 * 2 tasks
  const int token = task & (L_SEQ - 1);
  const int rest = task >> 11;               // 0..47
  const int head = rest % N_HEADS;
  const int isK = rest / N_HEADS;
  const float* wsc = isK ? k_scale : q_scale;
  const float* src = h + (size_t)token * W1_ROWS + ((isK ? N_HEADS : 0) + head) * HEAD_DIM;
  const int d = lane * 4;
  float4 t = *(const float4*)(src + d);
  float sq = t.x * t.x + t.y * t.y + t.z * t.z + t.w * t.w;
#pragma unroll
  for (int m = 1; m < 32; m <<= 1) sq += __shfl_xor(sq, m, 32);
  const float rs = rsqrtf(sq * (1.f / HEAD_DIM) + 1e-6f);
  float t0 = t.x * rs * wsc[d + 0];
  float t1 = t.y * rs * wsc[d + 1];
  float t2 = t.z * rs * wsc[d + 2];
  float t3 = t.w * rs * wsc[d + 3];
  // pe flat: [L][64][2][2]; [p][0] = {cos, -sin}, [p][1] = {sin, cos}
  const float* pp = pe + ((size_t)token * 64 + (d >> 1)) * 4;
  float o0 = pp[0] * t0 + pp[1] * t1;
  float o1 = pp[2] * t0 + pp[3] * t1;
  float o2 = pp[4] * t2 + pp[5] * t3;
  float o3 = pp[6] * t2 + pp[7] * t3;
  bf16* dst = (isK ? kb : qb) + ((size_t)head * L_SEQ + token) * HEAD_DIM + d;
  dst[0] = (bf16)o0; dst[1] = (bf16)o1; dst[2] = (bf16)o2; dst[3] = (bf16)o3;
}

// ---------------- 5) V transpose -> bf16 [head][128][L] ----------------------

__global__ __launch_bounds__(256) void vprep_kernel(const float* __restrict__ h,
                                                    unsigned short* __restrict__ vT_) {
  bf16* vT = (bf16*)vT_;
  __shared__ bf16 tile[32][129];
  const int head = blockIdx.y;
  const int t0 = blockIdx.x * 32;
  for (int i = threadIdx.x; i < 32 * 128; i += 256) {
    int r = i >> 7, c = i & 127;
    tile[r][c] = (bf16)h[(size_t)(t0 + r) * W1_ROWS + (48 + head) * HEAD_DIM + c];
  }
  __syncthreads();
  for (int i = threadIdx.x; i < 32 * 128; i += 256) {
    int c = i >> 5, r = i & 31;
    vT[((size_t)head * HEAD_DIM + c) * L_SEQ + t0 + r] = tile[r][c];
  }
}

// ---------------- 6) MLP silu(m1)*m2 -> concat[:, 3072:] ---------------------

__global__ __launch_bounds__(256) void mlpgate_kernel(const float* __restrict__ h,
                                                      unsigned short* __restrict__ cc_) {
  bf16* cc = (bf16*)cc_;
  const size_t idx = (size_t)blockIdx.x * 256 + threadIdx.x;
  if (idx >= (size_t)L_SEQ * MLP_DIM) return;
  const int row = (int)(idx / MLP_DIM);
  const int col = (int)(idx % MLP_DIM);
  const float* hr = h + (size_t)row * W1_ROWS + QKV_W;
  const float a = hr[col];
  const float b = hr[MLP_DIM + col];
  const float s = a / (1.f + __expf(-a));
  cc[(size_t)row * W2_COLS + DIM + col] = (bf16)(s * b);
}

// ---------------- 7) Flash attention -> concat[:, :3072] ---------------------

__global__ __launch_bounds__(128) void attn_kernel(const unsigned short* __restrict__ qb_,
                                                   const unsigned short* __restrict__ kb_,
                                                   const unsigned short* __restrict__ vT_,
                                                   unsigned short* __restrict__ cc_) {
  const bf16* qb = (const bf16*)qb_;
  const bf16* kb = (const bf16*)kb_;
  const bf16* vT = (const bf16*)vT_;
  bf16* cc = (bf16*)cc_;
  __shared__ bf16 plds[4][16][32];   // per-wave P tile
  const int head = blockIdx.x;
  const int wid = threadIdx.x >> 5, lane = threadIdx.x & 31;
  const int half = lane >> 4, cn = lane & 15;
  const int q0 = blockIdx.y * 64 + wid * 16;
  const bf16* qh = qb + (size_t)head * L_SEQ * HEAD_DIM;
  const bf16* kh = kb + (size_t)head * L_SEQ * HEAD_DIM;
  const bf16* vh = vT + (size_t)head * HEAD_DIM * L_SEQ;

  v16bf qa[4];
#pragma unroll
  for (int c = 0; c < 4; ++c) qa[c] = load_a16x32(qh, HEAD_DIM, q0, c * 32);

  v8f o[8];
  float mrun[8], lrun[8];
#pragma unroll
  for (int t = 0; t < 8; ++t) o[t] = zero_v8f();
#pragma unroll
  for (int j = 0; j < 8; ++j) { mrun[j] = -1e30f; lrun[j] = 0.f; }

  const float sc = 0.08838834764831845f;   // 1/sqrt(128)

  for (int kv = 0; kv < L_SEQ; kv += 32) {
    // Prefetch next K tile into the cache hierarchy (global_prefetch_b8).
    if (kv + 32 < L_SEQ)
      __builtin_prefetch(kh + (size_t)(kv + 32) * HEAD_DIM + lane * 8, 0, 1);
    // S = Q @ K^T over full head dim (4 x K=32 chunks), two 16-key fragments.
    v8f s0 = zero_v8f(), s1 = zero_v8f();
#pragma unroll
    for (int c = 0; c < 4; ++c) {
      v16bf b0 = load_b32x16(kh, HEAD_DIM, kv, c * 32);
      v16bf b1 = load_b32x16(kh, HEAD_DIM, kv + 16, c * 32);
      s0 = wmma_bf16(qa[c], b0, s0);
      s1 = wmma_bf16(qa[c], b1, s1);
    }
    // Online softmax; C-fragment rows: lanes 0-15 -> row j, lanes 16-31 -> row j+8.
#pragma unroll
    for (int j = 0; j < 8; ++j) {
      float a = s0[j] * sc, b = s1[j] * sc;
      float r = fmaxf(a, b);
      r = fmaxf(r, __shfl_xor(r, 1, 32));
      r = fmaxf(r, __shfl_xor(r, 2, 32));
      r = fmaxf(r, __shfl_xor(r, 4, 32));
      r = fmaxf(r, __shfl_xor(r, 8, 32));
      const float mnew = fmaxf(mrun[j], r);
      const float corr = __expf(mrun[j] - mnew);
      mrun[j] = mnew;
      const float p0 = __expf(a - mnew);
      const float p1 = __expf(b - mnew);
      s0[j] = p0;
      s1[j] = p1;
      float rs = p0 + p1;
      rs += __shfl_xor(rs, 1, 32);
      rs += __shfl_xor(rs, 2, 32);
      rs += __shfl_xor(rs, 4, 32);
      rs += __shfl_xor(rs, 8, 32);
      lrun[j] = lrun[j] * corr + rs;
#pragma unroll
      for (int t = 0; t < 8; ++t) o[t][j] *= corr;
    }
    // Re-layout P (C-fragment) -> A-fragment via per-wave LDS tile.
#pragma unroll
    for (int j = 0; j < 8; ++j) {
      plds[wid][j + half * 8][cn]      = (bf16)s0[j];
      plds[wid][j + half * 8][16 + cn] = (bf16)s1[j];
    }
    __syncthreads();
    v16bf pa = load_a16x32(&plds[wid][0][0], 32, 0, 0);
#pragma unroll
    for (int t = 0; t < 8; ++t) {
      v16bf vb = load_b32x16(vh, L_SEQ, t * 16, kv);
      o[t] = wmma_bf16(pa, vb, o[t]);
    }
    __syncthreads();
  }
  // Normalize and write bf16 into concat columns [head*128, head*128+128).
#pragma unroll
  for (int j = 0; j < 8; ++j) {
    const float inv = 1.f / lrun[j];
    const int row = q0 + j + half * 8;
#pragma unroll
    for (int t = 0; t < 8; ++t)
      cc[(size_t)row * W2_COLS + head * HEAD_DIM + t * 16 + cn] = (bf16)(o[t][j] * inv);
  }
}

// ---------------- host launcher ----------------------------------------------

extern "C" void kernel_launch(void* const* d_in, const int* in_sizes, int n_in,
                              void* d_out, int out_size, void* d_ws, size_t ws_size,
                              hipStream_t stream) {
  const float* x       = (const float*)d_in[0];
  const float* pe      = (const float*)d_in[1];
  const float* shift   = (const float*)d_in[2];
  const float* scale   = (const float*)d_in[3];
  const float* gate    = (const float*)d_in[4];
  const float* w1      = (const float*)d_in[5];
  const float* w2      = (const float*)d_in[6];
  const float* q_scale = (const float*)d_in[7];
  const float* k_scale = (const float*)d_in[8];
  float* out = (float*)d_out;

  char* ws = (char*)d_ws;
  size_t off = 0;
  auto take = [&](size_t bytes) -> void* {
    void* p = ws + off;
    off += (bytes + 255) & ~(size_t)255;
    return p;
  };
  unsigned short* xmod = (unsigned short*)take((size_t)L_SEQ * DIM * 2);
  unsigned short* w1b  = (unsigned short*)take((size_t)W1_ROWS * DIM * 2);
  unsigned short* w2b  = (unsigned short*)take((size_t)DIM * W2_COLS * 2);
  float*          h    = (float*)take((size_t)L_SEQ * W1_ROWS * 4);
  unsigned short* qb   = (unsigned short*)take((size_t)N_HEADS * L_SEQ * HEAD_DIM * 2);
  unsigned short* kb   = (unsigned short*)take((size_t)N_HEADS * L_SEQ * HEAD_DIM * 2);
  unsigned short* vT   = (unsigned short*)take((size_t)N_HEADS * HEAD_DIM * L_SEQ * 2);
  unsigned short* cc   = (unsigned short*)take((size_t)L_SEQ * W2_COLS * 2);

  // 1) weights -> bf16
  wcvt_kernel<<<8192, 256, 0, stream>>>(w1, w2, w1b, w2b);
  // 2) LN + modulation
  ln_mod_kernel<<<L_SEQ, 256, 0, stream>>>(x, shift, scale, xmod);
  // 3) fused QKV+MLP GEMM (TDM double-buffered, K-step 64)
  gemm1_kernel<<<dim3(W1_ROWS / 64, L_SEQ / 64), 128, 0, stream>>>(xmod, w1b, h);
  // 4) q/k RMSNorm + RoPE
  qkprep_kernel<<<(L_SEQ * N_HEADS * 2) / 4, 128, 0, stream>>>(h, pe, q_scale, k_scale, qb, kb);
  // 5) V transpose
  vprep_kernel<<<dim3(L_SEQ / 32, N_HEADS), 256, 0, stream>>>(h, vT);
  // 6) MLP gate
  mlpgate_kernel<<<(int)(((size_t)L_SEQ * MLP_DIM + 255) / 256), 256, 0, stream>>>(h, cc);
  // 7) flash attention
  attn_kernel<<<dim3(N_HEADS, L_SEQ / 64), 128, 0, stream>>>(qb, kb, vT, cc);
  // 8) output GEMM + gated residual (TDM double-buffered, K-step 64)
  gemm2_kernel<<<dim3(DIM / 64, L_SEQ / 64), 128, 0, stream>>>(cc, w2b, x, gate, out);
}
